// PGCNCritic_64905545777204
// MI455X (gfx1250) — matrix operand order repaired and checked by reading, not compiled, not run on hardware
//
#include <hip/hip_runtime.h>

typedef __attribute__((ext_vector_type(16))) _Float16 v16h;
typedef __attribute__((ext_vector_type(8)))  _Float16 v8h;
typedef __attribute__((ext_vector_type(8)))  float    v8f;

#define HID  64
#define NN   64
#define NDEV 63
#define FH   128

// padded LDS strides (halfs): chosen so 16-lane strided access spreads over banks
#define LDA 72    // k-stride for 64-wide arrays (36 dwords, gcd(36,64)=4 -> 16 banks)
#define LDE 40    // k-stride for 32-wide emb A/B (20 dwords)
#define LDF 200   // k-stride for 192-wide Wf1^T (100 dwords)

// ws (f16) layout, all B-side weights pre-TRANSPOSED (N-major) and pre-padded:
//  [0, 64*LDE)            w_emb^T   64 x LDE   (k<32 valid)
//  [WS_W1, +3*64*LDA)     W1^T,W2^T,W3^T  64 x LDA each (k<64 valid)
//  [WS_WF1, +128*LDF)     Wf1^T    128 x LDF  (k<192 valid)
#define WS_EMB 0
#define WS_W1  (64 * LDE)
#define WS_WF1 (WS_W1 + 3 * 64 * LDA)
#define WS_TOT (WS_WF1 + FH * LDF)

__device__ __forceinline__ v8f wmma_f16(v16h a, v16h b, v8f c) {
  return __builtin_amdgcn_wmma_f32_16x16x32_f16(false, a, false, b, (short)0, c, false, false);
}

// A fragment 16x32 f16 from row-major (M x K): lane&15 = row; two contiguous 8-half runs -> 2x b128
__device__ __forceinline__ v16h load_a(const _Float16* src, int ld, int row0, int k0, int lane) {
  const _Float16* p = src + (row0 + (lane & 15)) * ld + k0;
  int hi = (lane >> 4) << 3;
  v16h a;
#pragma unroll
  for (int j = 0; j < 8; ++j) a[j] = p[hi + j];
#pragma unroll
  for (int j = 0; j < 8; ++j) a[8 + j] = p[16 + hi + j];
  return a;
}

// A fragment with all rows identical (broadcast mean / server rows of the concat)
__device__ __forceinline__ v16h load_a_bcast(const _Float16* vec, int k0, int lane) {
  int hi = (lane >> 4) << 3;
  v16h a;
#pragma unroll
  for (int j = 0; j < 8; ++j) a[j] = vec[k0 + hi + j];
#pragma unroll
  for (int j = 0; j < 8; ++j) a[8 + j] = vec[k0 + 16 + hi + j];
  return a;
}

// B fragment 32x16 from TRANSPOSED source (N x K row-major): two contiguous 8-half runs -> 2x b128
__device__ __forceinline__ v16h load_bT(const _Float16* srcT, int ldT, int k0, int n0, int lane) {
  const _Float16* p = srcT + (n0 + (lane & 15)) * ldT + k0 + ((lane >> 4) << 3);
  v16h b;
#pragma unroll
  for (int j = 0; j < 16; ++j) b[j] = p[(j < 8) ? j : (16 + j - 8)];
  return b;
}

// ---------- prepass: weights -> f16, transposed (N-major) + padded, into workspace ----------
__global__ void pgcn_prep(const float* __restrict__ Wdev, const float* __restrict__ Wsrv,
                          const float* __restrict__ W1, const float* __restrict__ W2,
                          const float* __restrict__ W3, const float* __restrict__ Wf1,
                          _Float16* __restrict__ ws) {
  int i = blockIdx.x * blockDim.x + threadIdx.x;
  if (i >= WS_TOT) return;
  float v = 0.f;
  if (i < WS_W1) {                               // combined embedding weight^T  (n, k)
    int n = i / LDE, k = i % LDE;
    if (k < 14)      v = Wdev[k * 64 + n];
    else if (k < 17) v = Wsrv[(k - 14) * 64 + n];
  } else if (i < WS_WF1) {                       // W1..W3 ^T
    int j = i - WS_W1;
    int layer = j / (64 * LDA);
    int r = j % (64 * LDA);
    int n = r / LDA, k = r % LDA;
    if (k < 64) {
      const float* W = (layer == 0) ? W1 : ((layer == 1) ? W2 : W3);
      v = W[k * 64 + n];
    }
  } else {                                       // Wf1 ^T
    int j = i - WS_WF1;
    int n = j / LDF, k = j % LDF;
    if (k < 192) v = Wf1[k * FH + n];
  }
  ws[i] = (_Float16)v;
}

// ---------- main: one block per batch element, 4 waves ----------
__global__ __launch_bounds__(128) void pgcn_main(
    const float* __restrict__ dev_obs, const float* __restrict__ srv_obs,
    const float* __restrict__ adj,
    const float* __restrict__ b_dev, const float* __restrict__ b_srv,
    const float* __restrict__ b1, const float* __restrict__ b2, const float* __restrict__ b3,
    const float* __restrict__ bf1, const float* __restrict__ wf2, const float* __restrict__ bf2,
    const _Float16* __restrict__ ws, float* __restrict__ out)
{
  __shared__ __align__(16) _Float16 lds_adjn[NN * LDA];   // normalized adjacency (A-side)
  __shared__ __align__(16) _Float16 lds_h[NN * LDA];      // features, row-major (A-side)
  __shared__ __align__(16) _Float16 lds_hT[HID * LDA];    // features, transposed (B-side)
  __shared__ __align__(16) _Float16 lds_s[NN * LDA];      // scratch: adj stage / emb A / AH / reduce
  __shared__ __align__(16) _Float16 lds_w[HID * LDA];     // current layer weight^T
  __shared__ __align__(16) _Float16 lds_wf1[FH * LDF];    // Wf1^T resident (50KB)
  __shared__ float     lds_dis[NN];
  __shared__ _Float16  lds_mean[HID];

  const int tid  = threadIdx.x;
  const int lane = tid & 31;
  const int wave = tid >> 5;
  const int b    = blockIdx.x;
  const int row0 = wave * 16;
  const int rb   = row0 + ((lane >> 4) << 3);

  // ---- adjacency: b128 global loads -> padded f16 stage (exact for 0/1) ----
  const float4* Ag4 = (const float4*)(adj + (size_t)b * NN * NN);
  for (int i = tid; i < (NN * NN) / 4; i += 128) {
    float4 v = Ag4[i];
    int o = i * 4;
    _Float16* p = lds_s + (o >> 6) * LDA + (o & 63);
    p[0] = (_Float16)v.x; p[1] = (_Float16)v.y; p[2] = (_Float16)v.z; p[3] = (_Float16)v.w;
  }
  // stage Wf1^T once (coalesced b128 copy; flat layout identical in ws and LDS)
  {
    const int4* src4 = (const int4*)(ws + WS_WF1);
    int4* dst4 = (int4*)lds_wf1;
    for (int i = tid; i < (FH * LDF) / 8; i += 128) dst4[i] = src4[i];
  }
  __syncthreads();
  if (tid < NN) {
    float s = 0.f;
    for (int j = 0; j < NN; ++j) s += (float)lds_s[tid * LDA + j];
    lds_dis[tid] = rsqrtf(s < 1.f ? 1.f : s);
  }
  __syncthreads();
  for (int i = tid; i < (NN * NN) / 4; i += 128) {
    int o = i * 4;
    int r = o >> 6, c = o & 63;
    float dr = lds_dis[r];
    _Float16* ps = lds_s + r * LDA + c;
    _Float16* pa = lds_adjn + r * LDA + c;
#pragma unroll
    for (int j = 0; j < 4; ++j) pa[j] = (_Float16)(dr * lds_dis[c + j] * (float)ps[j]);
  }
  __syncthreads();

  // ---- embedding A (64 x 32, padded to LDE) in lds_s; w_emb^T into lds_w ----
  const float* dob = dev_obs + (size_t)b * NDEV * 14;
  const float* sob = srv_obs + (size_t)b * 3;
  for (int i = tid; i < NN * LDE; i += 128) {
    int r = i / LDE, k = i % LDE;
    float v = 0.f;
    if (r < NDEV) { if (k < 14) v = dob[r * 14 + k]; }
    else          { if (k >= 14 && k < 17) v = sob[k - 14]; }
    lds_s[i] = (_Float16)v;
  }
  {
    const int4* src4 = (const int4*)(ws + WS_EMB);
    int4* dst4 = (int4*)lds_w;
    for (int i = tid; i < (64 * LDE) / 8; i += 128) dst4[i] = src4[i];
  }
  __syncthreads();

  // ---- x = relu(A_emb @ W_emb + bias(row)) -> lds_h + lds_hT ----
  {
    v8f acc[4] = {};
    v16h a = load_a(lds_s, LDE, row0, 0, lane);
#pragma unroll
    for (int nt = 0; nt < 4; ++nt) {
      v16h bb = load_bT(lds_w, LDE, 0, nt * 16, lane);
      acc[nt] = wmma_f16(a, bb, acc[nt]);
    }
#pragma unroll
    for (int nt = 0; nt < 4; ++nt) {
      int col = nt * 16 + (lane & 15);
      float bd = b_dev[col], bs = b_srv[col];
      v8h t;
#pragma unroll
      for (int r = 0; r < 8; ++r) {
        int rr = rb + r;
        float v = acc[nt][r] + (rr < NDEV ? bd : bs);
        _Float16 hv = (_Float16)(v > 0.f ? v : 0.f);
        lds_h[rr * LDA + col] = hv;
        t[r] = hv;
      }
      *(v8h*)(lds_hT + col * LDA + rb) = t;   // transposed copy: one b128 store
    }
  }
  __syncthreads();

  // ---- 3 GCN layers: h = relu((adjn @ h) @ W + b) ----
  const float* lb[3] = { b1, b2, b3 };
#pragma unroll
  for (int layer = 0; layer < 3; ++layer) {
    {
      const int4* src4 = (const int4*)(ws + WS_W1 + layer * 64 * LDA);
      int4* dst4 = (int4*)lds_w;
      for (int i = tid; i < (64 * LDA) / 8; i += 128) dst4[i] = src4[i];
    }

    { // S = adjn @ h -> lds_s (A-side only, row-major)
      v8f acc[4] = {};
#pragma unroll
      for (int kt = 0; kt < 2; ++kt) {
        v16h a = load_a(lds_adjn, LDA, row0, kt * 32, lane);
#pragma unroll
        for (int nt = 0; nt < 4; ++nt) {
          v16h bb = load_bT(lds_hT, LDA, kt * 32, nt * 16, lane);
          acc[nt] = wmma_f16(a, bb, acc[nt]);
        }
      }
#pragma unroll
      for (int nt = 0; nt < 4; ++nt) {
        int col = nt * 16 + (lane & 15);
#pragma unroll
        for (int r = 0; r < 8; ++r) lds_s[(rb + r) * LDA + col] = (_Float16)acc[nt][r];
      }
    }
    __syncthreads();

    { // h = relu(S @ W + b) -> lds_h + lds_hT
      const float* bias = lb[layer];
      v8f acc[4] = {};
#pragma unroll
      for (int kt = 0; kt < 2; ++kt) {
        v16h a = load_a(lds_s, LDA, row0, kt * 32, lane);
#pragma unroll
        for (int nt = 0; nt < 4; ++nt) {
          v16h bb = load_bT(lds_w, LDA, kt * 32, nt * 16, lane);
          acc[nt] = wmma_f16(a, bb, acc[nt]);
        }
      }
#pragma unroll
      for (int nt = 0; nt < 4; ++nt) {
        int col = nt * 16 + (lane & 15);
        float bv = bias[col];
        v8h t;
#pragma unroll
        for (int r = 0; r < 8; ++r) {
          float v = acc[nt][r] + bv;
          _Float16 hv = (_Float16)(v > 0.f ? v : 0.f);
          lds_h[(rb + r) * LDA + col] = hv;
          t[r] = hv;
        }
        *(v8h*)(lds_hT + col * LDA + rb) = t;
      }
    }
    __syncthreads();
  }

  // ---- mean over device rows (contiguous along h^T rows) ----
  if (tid < HID) {
    float s = 0.f;
    for (int i = 0; i < NDEV; ++i) s += (float)lds_hT[tid * LDA + i];
    lds_mean[tid] = (_Float16)(s * (1.f / 63.f));
  }
  __syncthreads();

  // ---- critic: relu([h | mean | srv] @ Wf1 + bf1), concat via K-range source select ----
  v8f acc[8] = {};
#pragma unroll
  for (int ks = 0; ks < 6; ++ks) {
    int k0 = ks * 32;
    v16h a;
    if (k0 < 64)       a = load_a(lds_h, LDA, row0, k0, lane);
    else if (k0 < 128) a = load_a_bcast(lds_mean, k0 - 64, lane);
    else               a = load_a_bcast(lds_h + 63 * LDA, k0 - 128, lane);
#pragma unroll
    for (int nt = 0; nt < 8; ++nt) {
      v16h bb = load_bT(lds_wf1, LDF, k0, nt * 16, lane);
      acc[nt] = wmma_f16(a, bb, acc[nt]);
    }
  }
#pragma unroll
  for (int nt = 0; nt < 8; ++nt) {
    float bv = bf1[nt * 16 + (lane & 15)];
#pragma unroll
    for (int r = 0; r < 8; ++r) {
      float v = acc[nt][r] + bv;
      acc[nt][r] = v > 0.f ? v : 0.f;
    }
  }

  // ---- head: dot with Wf2 in registers + LDS cross-lane reduce ----
  float p[8];
#pragma unroll
  for (int r = 0; r < 8; ++r) p[r] = 0.f;
#pragma unroll
  for (int nt = 0; nt < 8; ++nt) {
    float w2 = wf2[nt * 16 + (lane & 15)];
#pragma unroll
    for (int r = 0; r < 8; ++r) p[r] += acc[nt][r] * w2;
  }
  float* red = (float*)lds_s;   // 64 rows x 16 partials = 4KB (fits in lds_s)
#pragma unroll
  for (int r = 0; r < 8; ++r) red[(rb + r) * 16 + (lane & 15)] = p[r];
  __syncthreads();
  if (tid < NDEV) {
    float s = bf2[0];
#pragma unroll
    for (int j = 0; j < 16; ++j) s += red[tid * 16 + j];
    out[(size_t)b * NDEV + tid] = s;
  }
}

extern "C" void kernel_launch(void* const* d_in, const int* in_sizes, int n_in,
                              void* d_out, int out_size, void* d_ws, size_t ws_size,
                              hipStream_t stream) {
  (void)in_sizes; (void)n_in; (void)out_size; (void)ws_size;
  const float* dev_obs = (const float*)d_in[0];
  const float* srv_obs = (const float*)d_in[1];
  const float* adjacency = (const float*)d_in[2];
  const float* W_dev = (const float*)d_in[3];
  const float* b_dev = (const float*)d_in[4];
  const float* W_srv = (const float*)d_in[5];
  const float* b_srv = (const float*)d_in[6];
  const float* W1 = (const float*)d_in[7];
  const float* b1 = (const float*)d_in[8];
  const float* W2 = (const float*)d_in[9];
  const float* b2 = (const float*)d_in[10];
  const float* W3 = (const float*)d_in[11];
  const float* b3 = (const float*)d_in[12];
  const float* Wf1 = (const float*)d_in[13];
  const float* bf1 = (const float*)d_in[14];
  const float* Wf2 = (const float*)d_in[15];
  const float* bf2 = (const float*)d_in[16];
  _Float16* ws = (_Float16*)d_ws;
  float* out = (float*)d_out;

  pgcn_prep<<<(WS_TOT + 255) / 256, 256, 0, stream>>>(W_dev, W_srv, W1, W2, W3, Wf1, ws);
  pgcn_main<<<16384, 128, 0, stream>>>(dev_obs, srv_obs, adjacency,
                                       b_dev, b_srv, b1, b2, b3,
                                       bf1, Wf2, bf2, ws, out);
}
